// SentenceTopKMoeBlock_44667659878790
// MI455X (gfx1250) — matrix operand order repaired and checked by reading, not compiled, not run on hardware
//
#include <hip/hip_runtime.h>
#include <hip/hip_bf16.h>
#include <math.h>

// Problem constants (from reference)
#define Bc 8
#define Sc 512
#define Hc 768
#define Ec 8
#define Fc 3072
#define Kc 2

typedef __attribute__((ext_vector_type(16))) __bf16 v16bf;
typedef __attribute__((ext_vector_type(8)))  float  v8f;

// Tiling
constexpr int TM     = 32;   // token rows per workgroup
constexpr int FCHUNK = 64;   // F columns per pipeline stage
constexpr int XP  = Hc + 8;      // padded X row in LDS (bf16 elems), 16B-aligned rows
constexpr int W1P = Hc + 8;      // W1s row [n_local_f][k=h]
constexpr int HSP = FCHUNK + 8;  // Hs row  [m][k=f_local]
constexpr int W2P = FCHUNK + 8;  // W2s row [n=h][k=f_local], 144B rows (16B-aligned)

constexpr size_t LDS_ELEMS = (size_t)TM * XP + (size_t)FCHUNK * W1P +
                             (size_t)TM * HSP + (size_t)Hc * W2P;
constexpr size_t LDS_BYTES = LDS_ELEMS * 2;   // ~258 KB of the 320 KB WGP LDS

// ---------------------------------------------------------------------------
// CDNA5 async global->LDS copies (ASYNCcnt path). The builtin takes pointers
// to 16-byte int vectors in explicit address spaces (per hipcc diagnostic):
//   (v4i AS1* src, v4i AS3* dst, imm offset, imm cpol)
// Falls back to a synchronous VGPR-passthrough copy if absent.
// ---------------------------------------------------------------------------
#if defined(__AMDGCN__) && defined(__gfx1250__) && \
    __has_builtin(__builtin_amdgcn_global_load_async_to_lds_b128)
#define ASYNC_LDS_COPY 1
typedef int v4i_ __attribute__((vector_size(4 * sizeof(int))));
#define AS1Q __attribute__((address_space(1)))
#define AS3Q __attribute__((address_space(3)))
#endif

__device__ __forceinline__ void lds_copy_b128(const __bf16* gsrc, __bf16* ldst) {
#ifdef ASYNC_LDS_COPY
    __builtin_amdgcn_global_load_async_to_lds_b128(
        (AS1Q v4i_*)gsrc, (AS3Q v4i_*)ldst, 0, 0);
#else
    *(uint4*)ldst = *(const uint4*)gsrc;
#endif
}

__device__ __forceinline__ void wait_async_copies() {
#ifdef ASYNC_LDS_COPY
#if __has_builtin(__builtin_amdgcn_s_wait_asynccnt)
    __builtin_amdgcn_s_wait_asynccnt(0);
#else
    asm volatile("s_wait_asynccnt 0" ::: "memory");
#endif
#endif
}

// ---------------------------------------------------------------------------
// fp32 -> bf16 straight conversion (hidden states)
// ---------------------------------------------------------------------------
__global__ void cvt_bf16_kernel(const float* __restrict__ src,
                                __bf16* __restrict__ dst, int n) {
    int i = blockIdx.x * blockDim.x + threadIdx.x;
    int stride = gridDim.x * blockDim.x;
    for (; i < n; i += stride) dst[i] = (__bf16)src[i];
}

// ---------------------------------------------------------------------------
// fp32 [e][R][C] -> bf16 [e][C][R] tiled transpose-convert (coalesced both
// sides). Makes all later LDS staging contiguous (async-copy eligible).
// Block: 32x8 threads, one 32x32 tile.
// ---------------------------------------------------------------------------
__global__ void transpose_cvt_kernel(const float* __restrict__ src,
                                     __bf16* __restrict__ dst, int R, int C) {
    __shared__ __bf16 tile[32][33];
    const int e = blockIdx.z;
    const float* s = src + (size_t)e * R * C;
    __bf16*      d = dst + (size_t)e * R * C;
    const int c0 = blockIdx.x * 32, r0 = blockIdx.y * 32;
    const int tx = threadIdx.x, ty = threadIdx.y;
    for (int i = ty; i < 32; i += 8)
        tile[i][tx] = (__bf16)s[(size_t)(r0 + i) * C + c0 + tx];
    __syncthreads();
    for (int i = ty; i < 32; i += 8)
        d[(size_t)(c0 + i) * R + r0 + tx] = tile[tx][i];
}

// ---------------------------------------------------------------------------
// Router: logits[b][e] = mean_s(hs[b,s,:]) . Wg[:,e]; softmax; top-2.
// ---------------------------------------------------------------------------
__global__ void router_kernel(const float* __restrict__ hs,
                              const float* __restrict__ Wg,
                              int*   __restrict__ sel,
                              float* __restrict__ selw,
                              float* __restrict__ logits_out) {
    __shared__ float pooled[Hc];
    __shared__ float lg[Ec];
    const int b = blockIdx.x;
    const int tid = threadIdx.x;

    for (int h = tid; h < Hc; h += blockDim.x) {
        float s = 0.f;
        const float* p = hs + (size_t)b * Sc * Hc + h;
        for (int ss = 0; ss < Sc; ++ss) s += p[(size_t)ss * Hc];
        pooled[h] = s;
    }
    __syncthreads();

    if (tid < Ec) {
        float d = 0.f;
        for (int h = 0; h < Hc; ++h) d += pooled[h] * Wg[h * Ec + tid];
        d *= (1.0f / (float)Sc);
        lg[tid] = d;
        logits_out[b * Ec + tid] = d;
    }
    __syncthreads();

    if (tid == 0) {
        float mx = lg[0];
        for (int e = 1; e < Ec; ++e) mx = fmaxf(mx, lg[e]);
        float sm[Ec]; float sum = 0.f;
        for (int e = 0; e < Ec; ++e) { sm[e] = __expf(lg[e] - mx); sum += sm[e]; }
        float inv = 1.0f / sum;
        for (int e = 0; e < Ec; ++e) sm[e] *= inv;
        int i0 = 0;
        for (int e = 1; e < Ec; ++e) if (sm[e] > sm[i0]) i0 = e;
        int i1 = (i0 == 0) ? 1 : 0;
        for (int e = 0; e < Ec; ++e) { if (e == i0) continue; if (sm[e] > sm[i1]) i1 = e; }
        sel[b * Kc + 0]  = i0;  sel[b * Kc + 1]  = i1;
        selw[b * Kc + 0] = sm[i0]; selw[b * Kc + 1] = sm[i1];
    }
}

// ---------------------------------------------------------------------------
// Fused top-K expert FFN. Pipeline per F-chunk (single-buffered LDS, async
// copies overlapped with WMMA):
//   issue W2s(i) async | GEMM1(i) on W1s(i) | wait+barrier |
//   issue W1s(i+1) async | GEMM2(i) on W2s(i) | wait+barrier
// ---------------------------------------------------------------------------
__global__ __launch_bounds__(256)
void moe_expert_kernel(const __bf16* __restrict__ hsb,   // [B][S][H]
                       const __bf16* __restrict__ w1t,   // [E][F][H] (transposed)
                       const __bf16* __restrict__ w2t,   // [E][H][F] (transposed)
                       const float*  __restrict__ b1,
                       const float*  __restrict__ b2,
                       const int*    __restrict__ sel,
                       const float*  __restrict__ selw,
                       float* __restrict__ out) {
    extern __shared__ __align__(16) char smem[];
    __bf16* Xs  = (__bf16*)smem;               // [TM][XP]
    __bf16* W1s = Xs  + (size_t)TM * XP;       // [FCHUNK][W1P]  ([f_local][h])
    __bf16* Hs  = W1s + (size_t)FCHUNK * W1P;  // [TM][HSP]
    __bf16* W2s = Hs  + (size_t)TM * HSP;      // [Hc][W2P]      ([h][f_local])

    const int b    = blockIdx.y;
    const int m0   = blockIdx.x * TM;
    const int tid  = threadIdx.x;
    const int lane = tid & 31;
    const int wave = tid >> 5;
    const int half = lane >> 4;
    const int l15  = lane & 15;
    const int wm   = wave & 1;
    const int wn   = wave >> 1;

    // X tile: 16B-chunk async copies of contiguous bf16 rows.
    {
        const __bf16* xrow = hsb + ((size_t)b * Sc + m0) * Hc;
        for (int idx = tid; idx < TM * (Hc / 8); idx += 256) {
            int r = idx / (Hc / 8), c8 = idx - r * (Hc / 8);
            lds_copy_b128(xrow + (size_t)r * Hc + c8 * 8, Xs + r * XP + c8 * 8);
        }
    }

    for (int kk = 0; kk < Kc; ++kk) {
        const int   e   = sel[b * Kc + kk];
        const float wgt = selw[b * Kc + kk];
        const __bf16* W1e = w1t + (size_t)e * Fc * Hc;   // [F][H]
        const __bf16* W2e = w2t + (size_t)e * Hc * Fc;   // [H][F]

        v8f acc[12];
        #pragma unroll
        for (int j = 0; j < 12; ++j) acc[j] = {};

        // Preload W1 chunk 0 (previous phase ended with a barrier).
        for (int idx = tid; idx < FCHUNK * (Hc / 8); idx += 256) {
            int n = idx / (Hc / 8), c8 = idx - n * (Hc / 8);
            lds_copy_b128(W1e + (size_t)n * Hc + c8 * 8, W1s + n * W1P + c8 * 8);
        }
        wait_async_copies();
        __syncthreads();   // Xs + W1s(0) ready

        for (int ic = 0; ic < Fc / FCHUNK; ++ic) {
            const int f0 = ic * FCHUNK;

            // Issue W2s(ic) async; it lands while GEMM1 runs on Xs/W1s.
            for (int idx = tid; idx < Hc * (FCHUNK / 8); idx += 256) {
                int h = idx / (FCHUNK / 8), c8 = idx - h * (FCHUNK / 8);
                lds_copy_b128(W2e + (size_t)h * Fc + f0 + c8 * 8,
                              W2s + h * W2P + c8 * 8);
            }

            // ---- GEMM1: Htile[32][64] = X[32][768] @ W1chunk ----
            v8f c1 = {};
            const int arow = wm * 16 + l15;
            const int nfl  = wn * 16 + l15;
            for (int kt = 0; kt < Hc; kt += 32) {
                v16bf af, bfr;
                #pragma unroll
                for (int v = 0; v < 8; ++v) {
                    int ka = kt + ((v < 4) ? 2 * v : 2 * v + 8) + half * 8;
                    af[2 * v]     = Xs[arow * XP + ka];
                    af[2 * v + 1] = Xs[arow * XP + ka + 1];
                    int kb = kt + ((v < 4) ? 2 * v : 2 * v + 8) + half * 8;
                    bfr[2 * v]     = W1s[nfl * W1P + kb];
                    bfr[2 * v + 1] = W1s[nfl * W1P + kb + 1];
                }
                c1 = __builtin_amdgcn_wmma_f32_16x16x32_bf16(false, af, false, bfr,
                                                             (short)0, c1, false, false);
            }
            // bias + exact GELU -> bf16 chunk in LDS
            const float bias1 = b1[(size_t)e * Fc + f0 + wn * 16 + l15];
            #pragma unroll
            for (int v = 0; v < 8; ++v) {
                float x = c1[v] + bias1;
                float g = 0.5f * x * (1.0f + erff(x * 0.70710678118654752f));
                int m = wm * 16 + v + half * 8;
                Hs[m * HSP + (wn * 16 + l15)] = (__bf16)g;
            }
            wait_async_copies();   // W2s(ic) landed (own wave's issues)
            __syncthreads();       // Hs visible, all waves' W2s done, W1s free

            // Issue W1s(ic+1) async; it lands while GEMM2 runs on Hs/W2s.
            if (ic + 1 < Fc / FCHUNK) {
                const __bf16* w1n = W1e + (size_t)(f0 + FCHUNK) * Hc;
                for (int idx = tid; idx < FCHUNK * (Hc / 8); idx += 256) {
                    int n = idx / (Hc / 8), c8 = idx - n * (Hc / 8);
                    lds_copy_b128(w1n + (size_t)n * Hc + c8 * 8,
                                  W1s + n * W1P + c8 * 8);
                }
            }

            // ---- GEMM2: acc[32][768] += Htile[32][64] @ W2chunk[64][768] ----
            #pragma unroll
            for (int kt = 0; kt < FCHUNK; kt += 32) {
                v16bf af;
                #pragma unroll
                for (int v = 0; v < 8; ++v) {
                    int ka = kt + ((v < 4) ? 2 * v : 2 * v + 8) + half * 8;
                    af[2 * v]     = Hs[arow * HSP + ka];
                    af[2 * v + 1] = Hs[arow * HSP + ka + 1];
                }
                #pragma unroll
                for (int j = 0; j < 12; ++j) {
                    const int h = (wn * 12 + j) * 16 + l15;
                    v16bf bfr;
                    #pragma unroll
                    for (int v = 0; v < 8; ++v) {
                        int kb = kt + ((v < 4) ? 2 * v : 2 * v + 8) + half * 8;
                        bfr[2 * v]     = W2s[h * W2P + kb];
                        bfr[2 * v + 1] = W2s[h * W2P + kb + 1];
                    }
                    acc[j] = __builtin_amdgcn_wmma_f32_16x16x32_bf16(false, af, false, bfr,
                                                                     (short)0, acc[j], false, false);
                }
            }
            wait_async_copies();   // W1s(ic+1) landed
            __syncthreads();       // all waves done reading Hs/W2s
        } // F chunks

        // Fold this expert into the output: expert 0 writes, expert 1 accumulates.
        #pragma unroll
        for (int j = 0; j < 12; ++j) {
            const int h = (wn * 12 + j) * 16 + l15;
            const float b2v = b2[(size_t)e * Hc + h];
            #pragma unroll
            for (int v = 0; v < 8; ++v) {
                const int s = m0 + wm * 16 + v + half * 8;
                const size_t o = ((size_t)b * Sc + s) * Hc + h;
                const float val = wgt * (acc[j][v] + b2v);
                if (kk == 0) out[o] = val; else out[o] += val;
            }
        }
    } // experts
}

// ---------------------------------------------------------------------------
extern "C" void kernel_launch(void* const* d_in, const int* in_sizes, int n_in,
                              void* d_out, int out_size, void* d_ws, size_t ws_size,
                              hipStream_t stream) {
    (void)in_sizes; (void)n_in; (void)out_size; (void)ws_size;
    const float* hs = (const float*)d_in[0];
    const float* Wg = (const float*)d_in[1];
    const float* W1 = (const float*)d_in[2];
    const float* b1 = (const float*)d_in[3];
    const float* W2 = (const float*)d_in[4];
    const float* b2 = (const float*)d_in[5];
    float* out = (float*)d_out;

    const size_t HS_N = (size_t)Bc * Sc * Hc;        // 3,145,728
    const size_t W_N  = (size_t)Ec * Hc * Fc;        // 18,874,368 each
    char* ws = (char*)d_ws;
    __bf16* hsb  = (__bf16*)(ws);
    __bf16* w1t  = (__bf16*)(ws + HS_N * 2);                 // [E][F][H]
    __bf16* w2t  = (__bf16*)(ws + HS_N * 2 + W_N * 2);       // [E][H][F]
    int*    sel  = (int*)  (ws + HS_N * 2 + 2 * W_N * 2);
    float*  selw = (float*)(ws + HS_N * 2 + 2 * W_N * 2 + 256);

    cvt_bf16_kernel<<<2048, 256, 0, stream>>>(hs, hsb, (int)HS_N);
    // W1 [E][H][F] -> w1t [E][F][H]
    transpose_cvt_kernel<<<dim3(Fc / 32, Hc / 32, Ec), dim3(32, 8), 0, stream>>>(W1, w1t, Hc, Fc);
    // W2 [E][F][H] -> w2t [E][H][F]
    transpose_cvt_kernel<<<dim3(Hc / 32, Fc / 32, Ec), dim3(32, 8), 0, stream>>>(W2, w2t, Fc, Hc);

    router_kernel<<<Bc, 256, 0, stream>>>(hs, Wg, sel, selw, out + HS_N);

    (void)hipFuncSetAttribute((const void*)moe_expert_kernel,
                              hipFuncAttributeMaxDynamicSharedMemorySize, (int)LDS_BYTES);

    dim3 grid(Sc / TM, Bc);
    moe_expert_kernel<<<grid, 256, LDS_BYTES, stream>>>(hsb, w1t, w2t, b1, b2,
                                                        sel, selw, out);
}
